// BlockMemory_1297080123751
// MI455X (gfx1250) — matrix operand order repaired and checked by reading, not compiled, not run on hardware
//
#include <hip/hip_runtime.h>

typedef __attribute__((ext_vector_type(16))) __bf16 v16bf;
typedef __attribute__((ext_vector_type(8)))  __bf16 v8bf;
typedef __attribute__((ext_vector_type(8)))  float  v8f;

#define CC 256
#define HH 64
#define WW 64
#define BB 16
#define SHRINK_L 0.0025f

// ---------- helpers ----------
__device__ __forceinline__ __bf16 f2bf(float f) {
  union { float f; unsigned u; } in; in.f = f;
  unsigned r = in.u + 0x7fffu + ((in.u >> 16) & 1u);   // round-to-nearest-even
  union { unsigned short s; __bf16 b; } out; out.s = (unsigned short)(r >> 16);
  return out.b;
}

__device__ __forceinline__ float wredsum(float v) {
#pragma unroll
  for (int o = 16; o > 0; o >>= 1) v += __shfl_xor(v, o, 32);
  return v;
}

// ---------- zero the 5 loss slots (graph-replay safe re-init) ----------
__global__ void bm_zero_loss(float* loss) {
  if (threadIdx.x < 5) loss[threadIdx.x] = 0.0f;
}

// ---------- NCHW input -> [T, C] NHWC tokens, with max-pool for blk<64 ----------
__global__ void bm_prepare(const float* __restrict__ x, float* __restrict__ x2,
                           int blk) {
  const int k = HH / blk;
  const long total = (long)BB * blk * blk * CC;
  for (long i = (long)blockIdx.x * blockDim.x + threadIdx.x; i < total;
       i += (long)gridDim.x * blockDim.x) {
    int c = (int)(i % CC);
    long t = i / CC;
    int xc = (int)(t % blk);
    int yr = (int)((t / blk) % blk);
    int b  = (int)(t / ((long)blk * blk));
    float m = -3.4e38f;
    for (int dy = 0; dy < k; ++dy)
      for (int dx = 0; dx < k; ++dx) {
        float v = x[(((long)b * CC + c) * HH + (yr * k + dy)) * WW + (xc * k + dx)];
        m = fmaxf(m, v);
      }
    x2[i] = m;
  }
}

// ---------- Wm f32 -> bf16 row-major [M,C] and transposed [C,M] ----------
__global__ void bm_cvtW(const float* __restrict__ Wm, __bf16* __restrict__ Wb,
                        __bf16* __restrict__ WbT, int M) {
  int i = blockIdx.x * blockDim.x + threadIdx.x;
  if (i < M * CC) {
    int c = i % CC, m = i / CC;
    __bf16 v = f2bf(Wm[i]);
    Wb[i] = v;
    WbT[(size_t)c * M + m] = v;
  }
}

// ---------- fused memory unit: GEMM1 -> softmax/top2/losses/shrink -> GEMM2 ----------
// Templated on M so both WMMA K-loops and all row passes fully unroll/pipeline.
template <int M>
__global__ __launch_bounds__(256) void bm_memunit(
    const float* __restrict__ x2f, const __bf16* __restrict__ Wb,
    const __bf16* __restrict__ WbT, const float* __restrict__ Wf,
    float* __restrict__ outp, float* __restrict__ loss, int blk,
    int chan_off, int direct, float invT) {
  extern __shared__ char smem_raw[];
  float*  xF   = (float*)smem_raw;                               // 16*256 f32
  __bf16* xA   = (__bf16*)(smem_raw + 16 * CC * 4);              // 16*256 bf16
  float*  att  = (float*)(smem_raw + 16 * CC * 6);               // 16*M  f32
  __bf16* attB = (__bf16*)(smem_raw + 16 * CC * 6 + 16 * M * 4); // 16*M  bf16

  const int tid  = threadIdx.x;
  const int lane = tid & 31;
  const int wave = tid >> 5;
  const long tbase = (long)blockIdx.x * 16;

  // load 16-token x tile (f32 + bf16 copies)
  for (int i = tid; i < 16 * CC; i += 256) {
    float v = x2f[tbase * CC + i];
    xF[i] = v;
    xA[i] = f2bf(v);
  }
  __syncthreads();

  const int koff = (lane >= 16) ? 8 : 0;     // unused; A fragments use boff
  const int boff = (lane >= 16) ? 16 : 0;    // A/B-fragment K offset (contiguous k = 16h + s)
  const int nb   = lane & 15;
  const int rrow = (lane >= 16) ? 8 : 0;     // C/D M offset (ISA layout)

  // ---- GEMM1: logits[16,M] = x2[16,256] @ Wm^T ----
  for (int mt = wave; mt < (M >> 4); mt += 8) {
    const int m_base = mt * 16;
    v8f acc = {};
#pragma unroll
    for (int kb = 0; kb < CC; kb += 32) {
      const __bf16* ap = &xA[(lane & 15) * CC + kb + boff];
      v8bf lo = *(const v8bf*)ap;
      v8bf hi = *(const v8bf*)(ap + 8);
      v16bf a = __builtin_shufflevector(lo, hi, 0, 1, 2, 3, 4, 5, 6, 7, 8, 9,
                                        10, 11, 12, 13, 14, 15);
      v16bf b = *(const v16bf*)&Wb[(size_t)(m_base + nb) * CC + kb + boff];
      acc = __builtin_amdgcn_wmma_f32_16x16x32_bf16(false, a, false, b,
                                                    (short)0, acc, false, false);
    }
#pragma unroll
    for (int r = 0; r < 8; ++r) att[(rrow + r) * M + m_base + nb] = acc[r];
  }
  __syncthreads();

  // ---- per-token: softmax, top-2, triplet/compact, shrink+renorm, entropy ----
  for (int rr = 0; rr < 2; ++rr) {
    const int row = wave * 2 + rr;
    float* L = &att[row * M];

    // top-2 of logits (== top-2 of softmax)
    float m1 = -3.4e38f, m2 = -3.4e38f;
    int i1 = 0, i2 = 0;
#pragma unroll 4
    for (int j = lane; j < M; j += 32) {
      float v = L[j];
      if (v > m1) { m2 = m1; i2 = i1; m1 = v; i1 = j; }
      else if (v > m2) { m2 = v; i2 = j; }
    }
#pragma unroll
    for (int off = 16; off > 0; off >>= 1) {
      float om1 = __shfl_xor(m1, off, 32), om2 = __shfl_xor(m2, off, 32);
      int   oi1 = __shfl_xor(i1, off, 32), oi2 = __shfl_xor(i2, off, 32);
      if (om1 > m1) {
        if (m1 >= om2) { m2 = m1; i2 = i1; } else { m2 = om2; i2 = oi2; }
        m1 = om1; i1 = oi1;
      } else if (om1 > m2) { m2 = om1; i2 = oi1; }
    }

    // softmax
    float s = 0.0f;
#pragma unroll 4
    for (int j = lane; j < M; j += 32) { float e = __expf(L[j] - m1); L[j] = e; s += e; }
    s = wredsum(s);
    const float inv = 1.0f / s;

    // triplet + compact from f32 data
    {
      const float* posr = &Wf[(size_t)i1 * CC];
      const float* negr = &Wf[(size_t)i2 * CC];
      const float* xr = &xF[row * CC];
      float dp = 0.0f, dn = 0.0f, cp = 0.0f;
#pragma unroll 4
      for (int c = lane; c < CC; c += 32) {
        float xv = xr[c];
        float a = xv - posr[c];
        float b = xv - negr[c];
        dp += (a + 1e-6f) * (a + 1e-6f);
        dn += (b + 1e-6f) * (b + 1e-6f);
        cp += a * a;
      }
      dp = wredsum(dp); dn = wredsum(dn); cp = wredsum(cp);
      if (lane == 0) {
        float tl = fmaxf(sqrtf(dp) - sqrtf(dn) + 1.0f, 0.0f);
        atomicAdd(&loss[1], tl * invT);                       // triplet
        atomicAdd(&loss[3], cp * invT * (1.0f / (float)CC));  // compact
      }
    }

    // hard-shrink-relu + L1 renorm + entropy (+ bf16 copy for GEMM2)
    float s2 = 0.0f;
#pragma unroll 4
    for (int j = lane; j < M; j += 32) {
      float a = L[j] * inv;
      float d = a - SHRINK_L;
      float shr = (d > 0.0f) ? d * a / (d + 1e-12f) : 0.0f;
      L[j] = shr; s2 += shr;
    }
    s2 = wredsum(s2);
    const float inv2 = 1.0f / fmaxf(s2, 1e-12f);
    float ent = 0.0f;
#pragma unroll 4
    for (int j = lane; j < M; j += 32) {
      float a = L[j] * inv2;
      attB[row * M + j] = f2bf(a);
      ent -= a * __logf(a + 1e-12f);
    }
    ent = wredsum(ent);
    if (lane == 0) atomicAdd(&loss[0], ent * invT);           // entropy
  }
  __syncthreads();

  // ---- GEMM2: rec[16,256] = att[16,M] @ Wm[M,256] (bf16 fragments from LDS) ----
  const int bb2 = blk * blk;
  for (int ct = wave; ct < 16; ct += 8) {
    const int c_base = ct * 16;
    v8f acc = {};
#pragma unroll
    for (int kb = 0; kb < M; kb += 32) {
      const __bf16* ap = &attB[(lane & 15) * M + kb + boff];
      v8bf lo = *(const v8bf*)ap;
      v8bf hi = *(const v8bf*)(ap + 8);
      v16bf a = __builtin_shufflevector(lo, hi, 0, 1, 2, 3, 4, 5, 6, 7, 8, 9,
                                        10, 11, 12, 13, 14, 15);
      v16bf b = *(const v16bf*)&WbT[(size_t)(c_base + nb) * M + kb + boff];
      acc = __builtin_amdgcn_wmma_f32_16x16x32_bf16(false, a, false, b,
                                                    (short)0, acc, false, false);
    }
    const int ch = c_base + nb;
#pragma unroll
    for (int r = 0; r < 8; ++r) {
      long tg = tbase + rrow + r;
      int b0 = (int)(tg / bb2);
      int rem = (int)(tg % bb2);
      int y = rem / blk, xq = rem % blk;
      float v = acc[r];
      if (direct)
        outp[(((long)b0 * 1024 + chan_off + ch) * HH + y) * WW + xq] = v;
      else
        outp[(((long)b0 * CC + ch) * blk + y) * blk + xq] = v;
    }
  }
}

// ---------- bilinear upsample blk -> 64 into concat slice ----------
__global__ void bm_resize(const float* __restrict__ rec, float* __restrict__ out,
                          int blk, int chan_off) {
  const long total = (long)BB * CC * HH * WW;
  const float scale = (float)blk / 64.0f;
  for (long i = (long)blockIdx.x * blockDim.x + threadIdx.x; i < total;
       i += (long)gridDim.x * blockDim.x) {
    int ox = (int)(i % WW);
    int oy = (int)((i / WW) % HH);
    int c  = (int)((i / (WW * HH)) % CC);
    int b  = (int)(i / ((long)WW * HH * CC));
    float sy = fminf(fmaxf((oy + 0.5f) * scale - 0.5f, 0.0f), blk - 1.0f);
    float sx = fminf(fmaxf((ox + 0.5f) * scale - 0.5f, 0.0f), blk - 1.0f);
    int y0 = (int)sy, x0 = (int)sx;
    int y1 = min(y0 + 1, blk - 1), x1 = min(x0 + 1, blk - 1);
    float fy = sy - y0, fx = sx - x0;
    const float* base = &rec[(((long)b * CC + c) * blk) * blk];
    float v00 = base[y0 * blk + x0], v01 = base[y0 * blk + x1];
    float v10 = base[y1 * blk + x0], v11 = base[y1 * blk + x1];
    float v = (1.0f - fy) * ((1.0f - fx) * v00 + fx * v01) +
              fy * ((1.0f - fx) * v10 + fx * v11);
    out[(((long)b * 1024 + chan_off + c) * HH + oy) * WW + ox] = v;
  }
}

// ---------- row norms (norm_loss) + squared norms cache ----------
__global__ void bm_norm(const float* __restrict__ Wf, float* __restrict__ sq,
                        float* __restrict__ loss, int M) {
  int row = blockIdx.x * 8 + (threadIdx.x >> 5);
  int lane = threadIdx.x & 31;
  if (row >= M) return;
  float s = 0.0f;
  for (int c = lane; c < CC; c += 32) {
    float v = Wf[(size_t)row * CC + c];
    s += v * v;
  }
  s = wredsum(s);
  if (lane == 0) {
    sq[row] = s;
    atomicAdd(&loss[2], fabsf(1.0f - sqrtf(s)) / (float)M);
  }
}

// ---------- pairwise distance_loss: sum_{i<j} relu(1 - d2) * 2/(M(M-1)) ----------
__global__ void bm_dist(const float* __restrict__ Wf, const float* __restrict__ sq,
                        float* __restrict__ loss, int M) {
  int row = blockIdx.x * 8 + (threadIdx.x >> 5);
  int lane = threadIdx.x & 31;
  if (row >= M) return;
  float xi[8];
#pragma unroll
  for (int t = 0; t < 8; ++t) xi[t] = Wf[(size_t)row * CC + lane + t * 32];
  const float sqi = sq[row];
  float acc = 0.0f;
  for (int j = row + 1; j < M; ++j) {
    const float* wj = &Wf[(size_t)j * CC];
    float d = 0.0f;
#pragma unroll
    for (int t = 0; t < 8; ++t) d += xi[t] * wj[lane + t * 32];
    d = wredsum(d);
    if (lane == 0) {
      float v = 1.0f - (sqi + sq[j] - 2.0f * d);
      if (v > 0.0f) acc += v;
    }
  }
  if (lane == 0)
    atomicAdd(&loss[4], acc * 2.0f / ((float)M * (float)(M - 1)));
}

// ---------- host launcher ----------
extern "C" void kernel_launch(void* const* d_in, const int* in_sizes, int n_in,
                              void* d_out, int out_size, void* d_ws, size_t ws_size,
                              hipStream_t stream) {
  (void)in_sizes; (void)n_in; (void)out_size; (void)ws_size;
  const float* x = (const float*)d_in[0];
  const float* W[4] = {(const float*)d_in[2], (const float*)d_in[3],
                       (const float*)d_in[4], (const float*)d_in[5]};
  const int Ms[4]   = {1024, 512, 256, 128};
  const int blks[4] = {64, 32, 16, 8};

  float* out = (float*)d_out;
  const long N = (long)BB * 1024 * HH * WW;  // 67108864
  float* loss = out + N;

  // workspace carve (sequential units -> buffers reused)
  char* w = (char*)d_ws;
  float* x2f = (float*)w;               w += (long)65536 * CC * 4;   // 64 MB
  __bf16* Wb  = (__bf16*)w;             w += (long)1024 * CC * 2;    // 512 KB
  __bf16* WbT = (__bf16*)w;             w += (long)1024 * CC * 2;    // 512 KB
  float* recbuf = (float*)w;            w += (long)BB * CC * 32 * 32 * 4; // 16 MB
  float* sq = (float*)w;

  bm_zero_loss<<<1, 32, 0, stream>>>(loss);

  for (int u = 0; u < 4; ++u) {
    const int blk = blks[u], M = Ms[u];
    const long T = (long)BB * blk * blk;

    bm_prepare<<<2048, 256, 0, stream>>>(x, x2f, blk);
    bm_cvtW<<<(M * CC + 255) / 256, 256, 0, stream>>>(W[u], Wb, WbT, M);

    const size_t smem =
        (size_t)16 * CC * 6 + (size_t)16 * M * 4 + (size_t)16 * M * 2;
    float* dst = (u == 0) ? out : recbuf;
    const int grid = (int)(T / 16);
    const int coff = 256 * u;
    const int direct = (u == 0) ? 1 : 0;
    const float invT = 1.0f / (float)T;

    switch (M) {
      case 1024:
        bm_memunit<1024><<<grid, 256, smem, stream>>>(x2f, Wb, WbT, W[u], dst,
                                                      loss, blk, coff, direct, invT);
        break;
      case 512:
        bm_memunit<512><<<grid, 256, smem, stream>>>(x2f, Wb, WbT, W[u], dst,
                                                     loss, blk, coff, direct, invT);
        break;
      case 256:
        bm_memunit<256><<<grid, 256, smem, stream>>>(x2f, Wb, WbT, W[u], dst,
                                                     loss, blk, coff, direct, invT);
        break;
      default:
        bm_memunit<128><<<grid, 256, smem, stream>>>(x2f, Wb, WbT, W[u], dst,
                                                     loss, blk, coff, direct, invT);
        break;
    }

    bm_norm<<<(M + 7) / 8, 256, 0, stream>>>(W[u], sq, loss, M);
    bm_dist<<<(M + 7) / 8, 256, 0, stream>>>(W[u], sq, loss, M);

    if (u > 0) bm_resize<<<4096, 256, 0, stream>>>(recbuf, out, blk, 256 * u);
  }
}
